// GarmentParamsPoint_4243427688563
// MI455X (gfx1250) — matrix operand order repaired and checked by reading, not compiled, not run on hardware
//
#include <hip/hip_runtime.h>
#include <math.h>

#define BN_EPS 1e-5f
#define INF_SENT 1e30f

typedef __attribute__((ext_vector_type(16))) _Float16 v16h;
typedef __attribute__((ext_vector_type(8)))  _Float16 v8h;
typedef __attribute__((ext_vector_type(8)))  float    v8f;

// ---------------------------------------------------------------------------
// WMMA fragment loaders (contiguous k order, same for A and B, wave32)
// A 16x32 f16: lane<16 row M=lane holds K{0..15}; lane>=16 holds K{16..31}
__device__ __forceinline__ v16h load_a_frag(const _Float16* base, int lda) {
  int lane = threadIdx.x & 31;
  int row  = lane & 15;
  int kb   = (lane & 16) ? 16 : 0;
  const _Float16* p = base + row * lda + kb;
  v8h lo = *(const v8h*)(p);
  v8h hi = *(const v8h*)(p + 8);
  v16h r;
#pragma unroll
  for (int i = 0; i < 8; ++i) { r[i] = lo[i]; r[i + 8] = hi[i]; }
  return r;
}

// B 32x16 f16 from transposed weights Wt[N][K]: lane<16 col N=lane holds K kk..kk+15,
// lane>=16 holds K kk+16..kk+31 (16 contiguous halves = one 32B load)
__device__ __forceinline__ v16h load_b_frag(const _Float16* wt, int ldb, int col, int kk) {
  int lane = threadIdx.x & 31;
  int kb   = kk + ((lane & 16) ? 16 : 0);
  return *(const v16h*)(wt + (size_t)col * ldb + kb);
}

// One fused Linear+ReLU+BN layer: A[Mtiles*16][lda] (LDS f16) x Wt[N][Kdim] -> H f16
__device__ __forceinline__ void gemm_layer(const _Float16* A, int lda,
                                           const _Float16* Wt, int Kdim,
                                           const float* bias, const float* bs, const float* bt,
                                           _Float16* H, int ldh, int Mtiles, int Ntiles) {
  int lane   = threadIdx.x & 31;
  int wave   = threadIdx.x >> 5;
  int nwaves = blockDim.x >> 5;
  int colL   = lane & 15;
  int rowOff = (lane & 16) ? 8 : 0;
  for (int tile = wave; tile < Mtiles * Ntiles; tile += nwaves) {
    int mt = tile / Ntiles, nt = tile % Ntiles;
    int col = nt * 16 + colL;
    float bv = bias[col];
    v8f c = { bv, bv, bv, bv, bv, bv, bv, bv };
    for (int kk = 0; kk < Kdim; kk += 32) {
      v16h a = load_a_frag(A + mt * 16 * lda + kk, lda);
      v16h b = load_b_frag(Wt, Kdim, col, kk);
      c = __builtin_amdgcn_wmma_f32_16x16x32_f16(false, a, false, b, (short)0, c, false, false);
    }
    float sv = bs[col], tv = bt[col];
#pragma unroll
    for (int r = 0; r < 8; ++r) {
      float v = fmaxf(c[r], 0.f) * sv + tv;
      H[(mt * 16 + rowOff + r) * ldh + col] = (_Float16)v;
    }
  }
}

// Final layer + masked max over rows [mt0*16, (mt0+nmt)*16); returns column max (all lanes)
__device__ __forceinline__ float colmax_tiles(const _Float16* A, int lda,
                                              const _Float16* Wt, int Kdim,
                                              const float* bias, const float* bs, const float* bt,
                                              int mt0, int nmt, int nt, int kcnt) {
  int lane   = threadIdx.x & 31;
  int colL   = lane & 15;
  int rowOff = (lane & 16) ? 8 : 0;
  int col = nt * 16 + colL;
  float bv = bias[col], sv = bs[col], tv = bt[col];
  float best = -INF_SENT;
  for (int i = 0; i < nmt; ++i) {
    int mt = mt0 + i;
    v8f c = { bv, bv, bv, bv, bv, bv, bv, bv };
    for (int kk = 0; kk < Kdim; kk += 32) {
      v16h a = load_a_frag(A + mt * 16 * lda + kk, lda);
      v16h b = load_b_frag(Wt, Kdim, col, kk);
      c = __builtin_amdgcn_wmma_f32_16x16x32_f16(false, a, false, b, (short)0, c, false, false);
    }
#pragma unroll
    for (int r = 0; r < 8; ++r) {
      int kidx = i * 16 + rowOff + r;
      if (kidx < kcnt) best = fmaxf(best, fmaxf(c[r], 0.f) * sv + tv);
    }
  }
  best = fmaxf(best, __shfl_xor(best, 16, 32));
  return best;
}

// ---------------------------------------------------------------------------
// Weight prep: W[Kin][N] f32 -> Wt[N][Kpad] f16 (zero pad); BN fold: s=g*rsqrt(rv+eps), t=bt-rm*s
__global__ void prep_layer(const float* W, const float* g, const float* bt,
                           const float* rm, const float* rv,
                           int Kin, int Kpad, int N, _Float16* wt, float* s, float* t) {
  int i = blockIdx.x * blockDim.x + threadIdx.x;
  if (i < N * Kpad) {
    int n = i / Kpad, k = i % Kpad;
    wt[(size_t)n * Kpad + k] = (k < Kin) ? (_Float16)W[(size_t)k * N + n] : (_Float16)0.f;
  }
  if (i < N) {
    float sv = g[i] * rsqrtf(rv[i] + BN_EPS);
    s[i] = sv;
    t[i] = bt[i] - rm[i] * sv;
  }
}

// ---------------------------------------------------------------------------
// Farthest point sampling: one block per batch; matches reference tie-break (first argmax)
__global__ void __launch_bounds__(256) fps_kernel(const float* pos, int N, int S, int* idx) {
  __shared__ float px[2048], py[2048], pz[2048], mind[2048];
  __shared__ float redv[256];
  __shared__ int   redi[256];
  int b = blockIdx.x;
  const float* p = pos + (size_t)b * N * 3;
  for (int n = threadIdx.x; n < N; n += 256) {
    px[n] = p[n * 3 + 0]; py[n] = p[n * 3 + 1]; pz[n] = p[n * 3 + 2];
  }
  __syncthreads();
  float cx = px[0], cy = py[0], cz = pz[0];
  for (int n = threadIdx.x; n < N; n += 256) {
    float dx = px[n] - cx, dy = py[n] - cy, dz = pz[n] - cz;
    mind[n] = dx * dx + dy * dy + dz * dz;
  }
  if (threadIdx.x == 0) idx[(size_t)b * S] = 0;
  __syncthreads();
  for (int it = 1; it < S; ++it) {
    float bv = -1.f; int bi = 0x7fffffff;
    for (int n = threadIdx.x; n < N; n += 256) {
      float v = mind[n];
      if (v > bv || (v == bv && n < bi)) { bv = v; bi = n; }
    }
    redv[threadIdx.x] = bv; redi[threadIdx.x] = bi;
    __syncthreads();
    for (int off = 128; off; off >>= 1) {
      if ((int)threadIdx.x < off) {
        float ov = redv[threadIdx.x + off]; int oi = redi[threadIdx.x + off];
        if (ov > redv[threadIdx.x] || (ov == redv[threadIdx.x] && oi < redi[threadIdx.x])) {
          redv[threadIdx.x] = ov; redi[threadIdx.x] = oi;
        }
      }
      __syncthreads();
    }
    int j = redi[0];
    if (threadIdx.x == 0) idx[(size_t)b * S + it] = j;
    float jx = px[j], jy = py[j], jz = pz[j];
    for (int n = threadIdx.x; n < N; n += 256) {
      float dx = px[n] - jx, dy = py[n] - jy, dz = pz[n] - jz;
      mind[n] = fminf(mind[n], dx * dx + dy * dy + dz * dz);
    }
    __syncthreads();
  }
}

// ---------------------------------------------------------------------------
// Radius neighbors, nearest-first (K rounds of wave argmin). 4 centroids/block (wave each).
__global__ void __launch_bounds__(128) nbr_kernel(const float* pos, const int* idx,
                                                  float r2, int N, int S,
                                                  int* nbr, int* cnt, float* cpos) {
  __shared__ float dd[4 * 2048];
  int wave = threadIdx.x >> 5, lane = threadIdx.x & 31;
  int bpb = S >> 2;
  int b = blockIdx.x / bpb;
  int sg = (size_t)b * S + (blockIdx.x % bpb) * 4 + wave;
  const float* pb = pos + (size_t)b * N * 3;
  int ci = idx[sg];
  float cx = pb[ci * 3 + 0], cy = pb[ci * 3 + 1], cz = pb[ci * 3 + 2];
  if (lane == 0) { cpos[sg * 3 + 0] = cx; cpos[sg * 3 + 1] = cy; cpos[sg * 3 + 2] = cz; }
  float* d = dd + wave * N;
  for (int n = lane; n < N; n += 32) {
    float dx = pb[n * 3 + 0] - cx, dy = pb[n * 3 + 1] - cy, dz = pb[n * 3 + 2] - cz;
    float v = dx * dx + dy * dy + dz * dz;
    d[n] = (v <= r2) ? v : INF_SENT;
  }
  int found = 32;
  for (int k = 0; k < 32; ++k) {
    float bv = INF_SENT; int bi = 0x7fffffff;
    for (int n = lane; n < N; n += 32) {
      float v = d[n];
      if (v < bv) { bv = v; bi = n; }
    }
    for (int off = 16; off; off >>= 1) {
      float ov = __shfl_xor(bv, off, 32); int oi = __shfl_xor(bi, off, 32);
      if (ov < bv || (ov == bv && oi < bi)) { bv = ov; bi = oi; }
    }
    if (bv >= INF_SENT) { found = k; break; }
    if (lane == 0) nbr[(size_t)sg * 32 + k] = bi;
    d[bi] = INF_SENT;  // same value from all lanes; in-order per-wave LDS
  }
  if (lane == 0) {
    cnt[sg] = found;
    for (int k = found; k < 32; ++k) nbr[(size_t)sg * 32 + k] = 0;
  }
}

// ---------------------------------------------------------------------------
// SA1 PointConv: 4 centroids/block (128 rows), 3->64->64->128, max over K=32
__global__ void __launch_bounds__(256)
sa1_conv(const float* pos, const int* nbr, const int* cnt, const float* cpos,
         const _Float16* w1, const float* b1, const float* s1, const float* t1,
         const _Float16* w2, const float* b2, const float* s2, const float* t2,
         const _Float16* w3, const float* b3, const float* s3, const float* t3,
         _Float16* fout) {
  __shared__ __align__(32) _Float16 sR0[128 * 64];  // A0 (128x32) then H2 (128x64)
  __shared__ __align__(32) _Float16 sH1[128 * 64];
  const int S = 1024, N = 2048;
  int bpb = S / 4;
  int b = blockIdx.x / bpb;
  int sgo = (blockIdx.x % bpb) * 4;
  const float* posb = pos + (size_t)b * N * 3;
  for (int row = threadIdx.x; row < 128; row += blockDim.x) {
    int sg = (size_t)b * S + sgo + (row >> 5);
    int nb = nbr[(size_t)sg * 32 + (row & 31)];
    _Float16* ar = sR0 + row * 32;
    ar[0] = (_Float16)(posb[nb * 3 + 0] - cpos[sg * 3 + 0]);
    ar[1] = (_Float16)(posb[nb * 3 + 1] - cpos[sg * 3 + 1]);
    ar[2] = (_Float16)(posb[nb * 3 + 2] - cpos[sg * 3 + 2]);
    for (int j = 3; j < 32; ++j) ar[j] = (_Float16)0.f;
  }
  __syncthreads();
  gemm_layer(sR0, 32, w1, 32, b1, s1, t1, sH1, 64, 8, 4);
  __syncthreads();
  gemm_layer(sH1, 64, w2, 64, b2, s2, t2, sR0, 64, 8, 4);
  __syncthreads();
  int wave = threadIdx.x >> 5;
  int c = wave & 3, grp = wave >> 2;
  int sg = (size_t)b * S + sgo + c;
  int kc = cnt[sg];
  for (int i = 0; i < 4; ++i) {
    int nt = grp * 4 + i;
    float m = colmax_tiles(sR0, 64, w3, 64, b3, s3, t3, 2 * c, 2, nt, kc);
    if ((threadIdx.x & 31) < 16)
      fout[(size_t)sg * 128 + nt * 16 + (threadIdx.x & 15)] = (_Float16)m;
  }
}

// SA2 PointConv: 2 centroids/block (64 rows), 160(pad of 131)->128->128->256
__global__ void __launch_bounds__(256)
sa2_conv(const float* pts, const _Float16* f1, const int* nbr, const int* cnt, const float* cen,
         const _Float16* w1, const float* b1, const float* s1, const float* t1,
         const _Float16* w2, const float* b2, const float* s2, const float* t2,
         const _Float16* w3, const float* b3, const float* s3, const float* t3,
         _Float16* fout) {
  __shared__ __align__(32) _Float16 sR0[64 * 160];  // A0 (64x160) then H2 (64x128)
  __shared__ __align__(32) _Float16 sH1[64 * 128];
  const int S = 256, N = 1024;
  int bpb = S / 2;
  int b = blockIdx.x / bpb;
  int sgo = (blockIdx.x % bpb) * 2;
  for (int t = threadIdx.x; t < 64 * 4; t += blockDim.x) {
    int row = t >> 2, q = t & 3;
    int sg = (size_t)b * S + sgo + (row >> 5);
    int nb = nbr[(size_t)sg * 32 + (row & 31)];
    const _Float16* fr = f1 + ((size_t)b * N + nb) * 128;
    _Float16* ar = sR0 + row * 160;
    for (int col = q * 40; col < q * 40 + 40; ++col) {
      _Float16 v;
      if (col < 128)      v = fr[col];
      else if (col < 131) v = (_Float16)(pts[((size_t)b * N + nb) * 3 + (col - 128)] -
                                         cen[(size_t)sg * 3 + (col - 128)]);
      else                v = (_Float16)0.f;
      ar[col] = v;
    }
  }
  __syncthreads();
  gemm_layer(sR0, 160, w1, 160, b1, s1, t1, sH1, 128, 4, 8);
  __syncthreads();
  gemm_layer(sH1, 128, w2, 128, b2, s2, t2, sR0, 128, 4, 8);
  __syncthreads();
  int wave = threadIdx.x >> 5;
  int c = wave & 1, grp = wave >> 1;
  int sg = (size_t)b * S + sgo + c;
  int kc = cnt[sg];
  for (int i = 0; i < 4; ++i) {
    int nt = grp * 4 + i;
    float m = colmax_tiles(sR0, 128, w3, 128, b3, s3, t3, 2 * c, 2, nt, kc);
    if ((threadIdx.x & 31) < 16)
      fout[(size_t)sg * 256 + nt * 16 + (threadIdx.x & 15)] = (_Float16)m;
  }
}

// MLP3: 32 rows/block (8 chunks per batch), 288(pad of 259)->256->512->1024, max over rows
__global__ void __launch_bounds__(256)
mlp3_kernel(const _Float16* f2, const float* cen,
            const _Float16* w1, const float* b1, const float* s1, const float* t1,
            const _Float16* w2, const float* b2, const float* s2, const float* t2,
            const _Float16* w3, const float* b3, const float* s3, const float* t3,
            float* partial) {
  __shared__ __align__(32) _Float16 sR0[32 * 512];  // A0 (32x288) then H2 (32x512)
  __shared__ __align__(32) _Float16 sH1[32 * 256];
  int b = blockIdx.x >> 3, ch = blockIdx.x & 7;
  for (int t = threadIdx.x; t < 32 * 8; t += blockDim.x) {
    int row = t >> 3, seg = t & 7;
    int g = b * 256 + ch * 32 + row;
    const _Float16* fr = f2 + (size_t)g * 256;
    _Float16* ar = sR0 + row * 288;
    for (int col = seg * 36; col < seg * 36 + 36; ++col) {
      _Float16 v;
      if (col < 256)      v = fr[col];
      else if (col < 259) v = (_Float16)cen[(size_t)g * 3 + (col - 256)];
      else                v = (_Float16)0.f;
      ar[col] = v;
    }
  }
  __syncthreads();
  gemm_layer(sR0, 288, w1, 288, b1, s1, t1, sH1, 256, 2, 16);
  __syncthreads();
  gemm_layer(sH1, 256, w2, 256, b2, s2, t2, sR0, 512, 2, 32);
  __syncthreads();
  int wave = threadIdx.x >> 5;
  for (int nt = wave; nt < 64; nt += 8) {
    float m = colmax_tiles(sR0, 512, w3, 512, b3, s3, t3, 0, 2, nt, 1 << 20);
    if ((threadIdx.x & 31) < 16)
      partial[((size_t)b * 8 + ch) * 1024 + nt * 16 + (threadIdx.x & 15)] = m;
  }
}

// Head: max over 8 chunks, 1024->512->256->12
__global__ void __launch_bounds__(256)
head_kernel(const float* partial, const float* W1, const float* bb1,
            const float* W2, const float* bb2, const float* W3, const float* bb3,
            float* out) {
  __shared__ float x[1024];
  __shared__ float y[512];
  int b = blockIdx.x;
  for (int j = threadIdx.x; j < 1024; j += 256) {
    float m = -INF_SENT;
    for (int ch = 0; ch < 8; ++ch) m = fmaxf(m, partial[((size_t)b * 8 + ch) * 1024 + j]);
    x[j] = m;
  }
  __syncthreads();
  for (int j = threadIdx.x; j < 512; j += 256) {
    float acc = bb1[j];
    for (int i = 0; i < 1024; ++i) acc += x[i] * W1[(size_t)i * 512 + j];
    y[j] = fmaxf(acc, 0.f);
  }
  __syncthreads();
  for (int j = threadIdx.x; j < 256; j += 256) {
    float acc = bb2[j];
    for (int i = 0; i < 512; ++i) acc += y[i] * W2[(size_t)i * 256 + j];
    x[j] = fmaxf(acc, 0.f);
  }
  __syncthreads();
  if (threadIdx.x < 12) {
    float acc = bb3[threadIdx.x];
    for (int i = 0; i < 256; ++i) acc += x[i] * W3[(size_t)i * 12 + threadIdx.x];
    out[(size_t)b * 12 + threadIdx.x] = acc;
  }
}

// ---------------------------------------------------------------------------
extern "C" void kernel_launch(void* const* d_in, const int* in_sizes, int n_in,
                              void* d_out, int out_size, void* d_ws, size_t ws_size,
                              hipStream_t stream) {
  (void)in_sizes; (void)n_in; (void)out_size; (void)ws_size;
  const float* positions = (const float*)d_in[0];

  char* ws = (char*)d_ws;
  size_t off = 0;
  auto take = [&](size_t bytes) -> void* {
    void* p = ws + off;
    off = (off + bytes + 255) & ~(size_t)255;
    return p;
  };

  // BN-folded f16 transposed weights + scale/shift
  struct L { _Float16* wt; float* s; float* t; int Kin; int Kpad; int N; int base; };
  L L1[3] = { { nullptr,nullptr,nullptr,   3,  32,   64,  1 },
              { nullptr,nullptr,nullptr,  64,  64,   64,  7 },
              { nullptr,nullptr,nullptr,  64,  64,  128, 13 } };
  L L2[3] = { { nullptr,nullptr,nullptr, 131, 160,  128, 19 },
              { nullptr,nullptr,nullptr, 128, 128,  128, 25 },
              { nullptr,nullptr,nullptr, 128, 128,  256, 31 } };
  L L3[3] = { { nullptr,nullptr,nullptr, 259, 288,  256, 37 },
              { nullptr,nullptr,nullptr, 256, 256,  512, 43 },
              { nullptr,nullptr,nullptr, 512, 512, 1024, 49 } };
  L* all[3] = { L1, L2, L3 };
  for (int m = 0; m < 3; ++m)
    for (int l = 0; l < 3; ++l) {
      L& e = all[m][l];
      e.wt = (_Float16*)take((size_t)e.N * e.Kpad * sizeof(_Float16));
      e.s  = (float*)take((size_t)e.N * sizeof(float));
      e.t  = (float*)take((size_t)e.N * sizeof(float));
    }

  int*   idx1  = (int*)take(16 * 1024 * 4);
  int*   cnt1  = (int*)take(16 * 1024 * 4);
  int*   nbr1  = (int*)take((size_t)16 * 1024 * 32 * 4);
  float* cpos1 = (float*)take((size_t)16 * 1024 * 3 * 4);
  _Float16* f1 = (_Float16*)take((size_t)16 * 1024 * 128 * 2);
  int*   idx2  = (int*)take(16 * 256 * 4);
  int*   cnt2  = (int*)take(16 * 256 * 4);
  int*   nbr2  = (int*)take((size_t)16 * 256 * 32 * 4);
  float* cpos2 = (float*)take((size_t)16 * 256 * 3 * 4);
  _Float16* f2 = (_Float16*)take((size_t)16 * 256 * 256 * 2);
  float* partial = (float*)take((size_t)16 * 8 * 1024 * 4);

  // 1) weight prep (9 layers)
  for (int m = 0; m < 3; ++m)
    for (int l = 0; l < 3; ++l) {
      L& e = all[m][l];
      int total = e.N * e.Kpad;
      prep_layer<<<(total + 255) / 256, 256, 0, stream>>>(
          (const float*)d_in[e.base], (const float*)d_in[e.base + 2],
          (const float*)d_in[e.base + 3], (const float*)d_in[e.base + 4],
          (const float*)d_in[e.base + 5], e.Kin, e.Kpad, e.N, e.wt, e.s, e.t);
    }

  // 2) SA1: FPS 2048->1024, radius 10, PointConv
  fps_kernel<<<16, 256, 0, stream>>>(positions, 2048, 1024, idx1);
  nbr_kernel<<<16 * (1024 / 4), 128, 0, stream>>>(positions, idx1, 100.f, 2048, 1024,
                                                  nbr1, cnt1, cpos1);
  sa1_conv<<<16 * (1024 / 4), 256, 0, stream>>>(
      positions, nbr1, cnt1, cpos1,
      L1[0].wt, (const float*)d_in[2],  L1[0].s, L1[0].t,
      L1[1].wt, (const float*)d_in[8],  L1[1].s, L1[1].t,
      L1[2].wt, (const float*)d_in[14], L1[2].s, L1[2].t, f1);

  // 3) SA2: FPS 1024->256, radius 40, PointConv on [f1|rel]
  fps_kernel<<<16, 256, 0, stream>>>(cpos1, 1024, 256, idx2);
  nbr_kernel<<<16 * (256 / 4), 128, 0, stream>>>(cpos1, idx2, 1600.f, 1024, 256,
                                                 nbr2, cnt2, cpos2);
  sa2_conv<<<16 * (256 / 2), 256, 0, stream>>>(
      cpos1, f1, nbr2, cnt2, cpos2,
      L2[0].wt, (const float*)d_in[20], L2[0].s, L2[0].t,
      L2[1].wt, (const float*)d_in[26], L2[1].s, L2[1].t,
      L2[2].wt, (const float*)d_in[32], L2[2].s, L2[2].t, f2);

  // 4) MLP3 on [f2|p2] + partial row-max
  mlp3_kernel<<<16 * 8, 256, 0, stream>>>(
      f2, cpos2,
      L3[0].wt, (const float*)d_in[38], L3[0].s, L3[0].t,
      L3[1].wt, (const float*)d_in[44], L3[1].s, L3[1].t,
      L3[2].wt, (const float*)d_in[50], L3[2].s, L3[2].t, partial);

  // 5) head: global max + lin1/lin2/lin3
  head_kernel<<<16, 256, 0, stream>>>(partial,
      (const float*)d_in[55], (const float*)d_in[56],
      (const float*)d_in[57], (const float*)d_in[58],
      (const float*)d_in[59], (const float*)d_in[60],
      (float*)d_out);
}
